// BiLSTM_centric_layer_29592324669751
// MI455X (gfx1250) — compile-verified, matrix operand order checked by reading, not run on hardware
//
#include <hip/hip_runtime.h>
#include <math.h>

// ---------------------------------------------------------------- types ----
typedef __attribute__((ext_vector_type(16))) __bf16 v16bf;
typedef __attribute__((ext_vector_type(8)))  float  v8f;
typedef __attribute__((ext_vector_type(4)))  unsigned int u32x4;
typedef __attribute__((ext_vector_type(8)))  int  i32x8;
typedef __attribute__((ext_vector_type(4)))  int  i32x4;

union Frag {
    v16bf v;
    uint4 q[2];
};

__device__ inline v8f wmma_bf16(const Frag& a, const Frag& b, v8f c) {
    return __builtin_amdgcn_wmma_f32_16x16x32_bf16(
        /*neg_a=*/false, a.v, /*neg_b=*/false, b.v,
        /*c_mod=*/(short)0, c, /*reuse_a=*/false, /*reuse_b=*/false);
}

__device__ inline unsigned short f2bf(float f) {
    union { float f; unsigned int u; } x; x.f = f;
    unsigned int u = x.u;
    unsigned int r = (u + 0x7FFFu + ((u >> 16) & 1u)) >> 16;   // RNE
    return (unsigned short)r;
}

__device__ inline float sigm(float x) { return 1.0f / (1.0f + expf(-x)); }

// TDM: async 2D tile load (bf16 elements) global -> LDS.
// Tile is tile_rows x 32 elements, row stride in the global tensor is
// row_stride elements. D# built per CDNA5 ISA ch.8 (group0/group1 bitfields).
__device__ inline void tdm_load_2d(unsigned lds_addr,
                                   const unsigned short* gptr,
                                   unsigned tile_rows,
                                   unsigned row_stride) {
    unsigned long long ga = (unsigned long long)(size_t)gptr;
    u32x4 g0 = {
        1u,                                    // count=1, user descriptor
        lds_addr,                              // LDS byte address
        (unsigned)(ga & 0xFFFFFFFFu),          // global addr [31:0]
        (unsigned)((ga >> 32) & 0x01FFFFFFu) | 0x80000000u  // [56:32] | type=2<<30
    };
    i32x8 g1 = {
        (int)(1u << 16),                                   // data_size=1 (2 bytes)
        (int)((32u & 0xFFFFu) << 16),                      // tensor_dim0=32 (lo16)
        (int)(((32u >> 16) & 0xFFFFu) | ((tile_rows & 0xFFFFu) << 16)), // dim0 hi | dim1 lo
        (int)(((tile_rows >> 16) & 0xFFFFu) | ((32u & 0xFFFFu) << 16)), // dim1 hi | tile_dim0=32
        (int)(tile_rows & 0xFFFFu),                        // tile_dim1
        (int)(row_stride),                                 // dim0_stride [31:0]
        (int)((((unsigned long long)row_stride) >> 32) & 0xFFFFu),
        0
    };
    i32x4 zz4 = {0, 0, 0, 0};
    i32x8 zz8 = {0, 0, 0, 0, 0, 0, 0, 0};
    // 6-arg form on this toolchain: (g0, g1, g2, g3, g4, cpol)
    __builtin_amdgcn_tensor_load_to_lds(g0, g1, zz4, zz4, zz8, 0);
}

// Problem constants
#define BB 32
#define S_RAW 1024
#define S_SUM 128
#define D_IN 300
#define KPAD_IN 320
#define HH 256
#define G4H 1024   /* 4*H */
#define TWO_H 512
#define NHEAD 4
#define DHEAD 128

// --------------------------------------------------------- pack kernels ----
__global__ void pack_bf16_pad(const float* __restrict__ src,
                              unsigned short* __restrict__ dst,
                              int rows, int K, int Kpad) {
    int idx = blockIdx.x * blockDim.x + threadIdx.x;
    int total = rows * Kpad;
    if (idx >= total) return;
    int r = idx / Kpad, k = idx - r * Kpad;
    float v = (k < K) ? src[(size_t)r * K + k] : 0.0f;
    dst[idx] = f2bf(v);
}

__global__ void pack_wq(const float* __restrict__ Wq, unsigned short* __restrict__ dst) {
    int idx = blockIdx.x * blockDim.x + threadIdx.x;   // 512*512
    if (idx >= TWO_H * TWO_H) return;
    int n = idx / TWO_H, d = idx - n * TWO_H;
    int h = n / DHEAD, e = n - h * DHEAD;
    dst[idx] = f2bf(Wq[((size_t)h * TWO_H + d) * DHEAD + e]);
}

// ---------------------------------------------------------- WMMA GEMM -----
// C[M,N] = A[M,K](bf16) * W[N,K](bf16)^T (+bias). K mult of 32, M mult of 128,
// N mult of 64. 256 threads = 8 waves; wave w owns M rows [16w,16w+16) and all
// 4 N tiles. A/B tiles staged by the Tensor Data Mover, double buffered.
__global__ __launch_bounds__(256) void gemm_bf16(
        const unsigned short* __restrict__ A,
        const unsigned short* __restrict__ W,
        const float* __restrict__ bias,
        float* __restrict__ C,
        int M, int N, int K) {
    __shared__ __align__(128) unsigned short As[2][128 * 32];  // 2 x 8KB
    __shared__ __align__(128) unsigned short Bs[2][64 * 32];   // 2 x 4KB
    const int tid  = threadIdx.x;
    const int wave = tid >> 5;          // 0..7 : M tile
    const int lane = tid & 31;
    const int l15  = lane & 15;
    const int hi   = lane >> 4;
    const int koff = hi ? 8 : 0;
    const int mBase = blockIdx.x * 128;
    const int nBase = blockIdx.y * 64;

    v8f acc[4] = {};

    if (wave == 0) {   // TDM is per-wave; issue from wave 0 only (EXEC ignored)
        tdm_load_2d((unsigned)(size_t)&As[0][0], A + (size_t)mBase * K, 128, (unsigned)K);
        tdm_load_2d((unsigned)(size_t)&Bs[0][0], W + (size_t)nBase * K, 64, (unsigned)K);
        __builtin_amdgcn_s_wait_tensorcnt(0);
    }
    __syncthreads();

    int buf = 0;
    for (int k0 = 0; k0 < K; k0 += 32) {
        if (wave == 0 && (k0 + 32) < K) {   // overlap DMA of next tile with WMMA
            tdm_load_2d((unsigned)(size_t)&As[buf ^ 1][0],
                        A + (size_t)mBase * K + (k0 + 32), 128, (unsigned)K);
            tdm_load_2d((unsigned)(size_t)&Bs[buf ^ 1][0],
                        W + (size_t)nBase * K + (k0 + 32), 64, (unsigned)K);
        }
        Frag fa;
        const int arow = wave * 16 + l15;
        fa.q[0] = *(const uint4*)(&As[buf][arow * 32 + koff]);
        fa.q[1] = *(const uint4*)(&As[buf][arow * 32 + koff + 16]);
#pragma unroll
        for (int nt = 0; nt < 4; ++nt) {
            Frag fb;
            const int bcol = nt * 16 + l15;
            fb.q[0] = *(const uint4*)(&Bs[buf][bcol * 32 + koff]);
            fb.q[1] = *(const uint4*)(&Bs[buf][bcol * 32 + koff + 16]);
            acc[nt] = wmma_bf16(fa, fb, acc[nt]);
        }
        if (wave == 0) __builtin_amdgcn_s_wait_tensorcnt(0);
        __syncthreads();
        buf ^= 1;
    }

#pragma unroll
    for (int nt = 0; nt < 4; ++nt) {
        int col = nBase + nt * 16 + l15;
        float bv = bias ? bias[col] : 0.0f;
#pragma unroll
        for (int r = 0; r < 8; ++r) {
            int row = mBase + wave * 16 + r + hi * 8;
            C[(size_t)row * N + col] = acc[nt][r] + bv;
        }
    }
}

// ----------------------------------------------------- LSTM recurrence ----
struct LstmCfg {
    const float* xg;              // [B, S, 4H] fp32 (bias pre-added)
    const unsigned short* whh;    // [4H, 256] bf16 row-major
    float* out;                   // [B, S, 512] fp32
    int S;
    int reverse;
    int colOff;                   // 0 fwd, 256 bwd
};

__global__ __launch_bounds__(1024) void lstm_scan(LstmCfg c0, LstmCfg c1,
                                                  LstmCfg c2, LstmCfg c3) {
    LstmCfg cfg = c0;
    if (blockIdx.x == 1) cfg = c1;
    else if (blockIdx.x == 2) cfg = c2;
    else if (blockIdx.x == 3) cfg = c3;

    extern __shared__ char smem[];
    unsigned short* h_bf = (unsigned short*)smem;            // 32*256 bf16 = 16KB
    float* c_st  = (float*)(smem + 16 * 1024);               // 32*256 f32  = 32KB
    float* gates = (float*)(smem + 48 * 1024);               // 32*1024 f32 = 128KB

    const int tid  = threadIdx.x;
    const int wave = tid >> 5;
    const int lane = tid & 31;
    const int l15  = lane & 15;
    const int hi   = lane >> 4;
    const int koff = hi ? 8 : 0;

    for (int i = tid; i < 32 * 256; i += 1024) { h_bf[i] = 0; c_st[i] = 0.0f; }
    __syncthreads();

    // Register-resident Whh fragments for this wave's 32 gate columns.
    Frag bf[2][8];
#pragma unroll
    for (int nj = 0; nj < 2; ++nj) {
        int n = wave * 32 + nj * 16 + l15;
#pragma unroll
        for (int kc = 0; kc < 8; ++kc) {
            int k0 = kc * 32 + koff;
            bf[nj][kc].q[0] = *(const uint4*)(cfg.whh + (size_t)n * 256 + k0);
            bf[nj][kc].q[1] = *(const uint4*)(cfg.whh + (size_t)n * 256 + k0 + 16);
        }
    }

    const int S = cfg.S;
    for (int t = 0; t < S; ++t) {
        const int tt = cfg.reverse ? (S - 1 - t) : t;
        if (t + 1 < S) {   // prefetch next timestep's xg slice
            const int tn = cfg.reverse ? (S - 2 - t) : (t + 1);
            __builtin_prefetch(&cfg.xg[((size_t)(l15) * S + tn) * G4H + wave * 32], 0, 0);
        }

        v8f acc[2][2];
#pragma unroll
        for (int mi = 0; mi < 2; ++mi)
#pragma unroll
            for (int nj = 0; nj < 2; ++nj) {
                int colg = wave * 32 + nj * 16 + l15;
#pragma unroll
                for (int r = 0; r < 8; ++r) {
                    int b = mi * 16 + r + hi * 8;
                    acc[mi][nj][r] = cfg.xg[((size_t)b * S + tt) * G4H + colg];
                }
            }
#pragma unroll
        for (int kc = 0; kc < 8; ++kc) {
            Frag a0, a1;
            int k0 = kc * 32 + koff;
            a0.q[0] = *(const uint4*)(&h_bf[(size_t)l15 * 256 + k0]);
            a0.q[1] = *(const uint4*)(&h_bf[(size_t)l15 * 256 + k0 + 16]);
            a1.q[0] = *(const uint4*)(&h_bf[(size_t)(l15 + 16) * 256 + k0]);
            a1.q[1] = *(const uint4*)(&h_bf[(size_t)(l15 + 16) * 256 + k0 + 16]);
            acc[0][0] = wmma_bf16(a0, bf[0][kc], acc[0][0]);
            acc[0][1] = wmma_bf16(a0, bf[1][kc], acc[0][1]);
            acc[1][0] = wmma_bf16(a1, bf[0][kc], acc[1][0]);
            acc[1][1] = wmma_bf16(a1, bf[1][kc], acc[1][1]);
        }
#pragma unroll
        for (int mi = 0; mi < 2; ++mi)
#pragma unroll
            for (int nj = 0; nj < 2; ++nj) {
                int colg = wave * 32 + nj * 16 + l15;
#pragma unroll
                for (int r = 0; r < 8; ++r) {
                    int b = mi * 16 + r + hi * 8;
                    gates[b * G4H + colg] = acc[mi][nj][r];
                }
            }
        __syncthreads();

#pragma unroll
        for (int u = 0; u < 8; ++u) {
            int idx = tid * 8 + u;             // 0 .. 8191
            int b = idx >> 8;
            int j = idx & 255;
            float gi = gates[b * G4H + j];
            float gf = gates[b * G4H + 256 + j];
            float gg = gates[b * G4H + 512 + j];
            float go = gates[b * G4H + 768 + j];
            float cn = sigm(gf) * c_st[b * 256 + j] + sigm(gi) * tanhf(gg);
            float hn = sigm(go) * tanhf(cn);
            c_st[b * 256 + j] = cn;
            h_bf[b * 256 + j] = f2bf(hn);
            cfg.out[((size_t)b * S + tt) * TWO_H + cfg.colOff + j] = hn;
        }
        __syncthreads();
    }
}

// ----------------------------------------------------------- epilogues ----
__global__ void meanpool(const float* __restrict__ out_sum,
                         const int* __restrict__ len_sum,
                         float* __restrict__ sum_vec) {
    int b = blockIdx.x;
    int d = threadIdx.x;
    int len = len_sum[b]; if (len < 1) len = 1;
    float s = 0.0f;
    for (int t = 0; t < len; ++t) s += out_sum[((size_t)b * S_SUM + t) * TWO_H + d];
    sum_vec[b * TWO_H + d] = s / (float)len;
}

__global__ void kvproj(const float* __restrict__ sum_vec,
                       const float* __restrict__ Wk,
                       const float* __restrict__ Wv,
                       float* __restrict__ kbuf, float* __restrict__ vbuf) {
    int idx = blockIdx.x * blockDim.x + threadIdx.x;
    if (idx >= 2 * NHEAD * BB * DHEAD) return;
    int which = idx >> 14;
    int r = idx & 16383;
    int h = r >> 12;
    int b = (r >> 7) & 31;
    int e = r & 127;
    const float* Wm = which ? Wv : Wk;
    float s = 0.0f;
    for (int d = 0; d < TWO_H; ++d)
        s += sum_vec[b * TWO_H + d] * Wm[((size_t)h * TWO_H + d) * DHEAD + e];
    (which ? vbuf : kbuf)[(h * BB + b) * DHEAD + e] = s;
}

__global__ void scoreskern(const float* __restrict__ q,
                           const float* __restrict__ kbuf,
                           float* __restrict__ scores) {
    int idx = blockIdx.x * blockDim.x + threadIdx.x;
    if (idx >= NHEAD * BB * S_RAW) return;
    int h = idx >> 15;
    int b = (idx >> 10) & 31;
    int s = idx & 1023;
    const float* qp = q + ((size_t)b * S_RAW + s) * TWO_H + h * DHEAD;
    const float* kp = kbuf + (h * BB + b) * DHEAD;
    float a = 0.0f;
    for (int e = 0; e < DHEAD; ++e) a += qp[e] * kp[e];
    scores[((size_t)h * BB + b) * S_RAW + s] = a;
}

__global__ void softmaxkern(const float* __restrict__ scores,
                            float* __restrict__ attn) {
    __shared__ float red[256];
    int row = blockIdx.x;
    const float* src = scores + (size_t)row * S_RAW;
    float m = -3.4e38f;
    for (int i = threadIdx.x; i < S_RAW; i += 256) m = fmaxf(m, src[i]);
    red[threadIdx.x] = m; __syncthreads();
    for (int s = 128; s > 0; s >>= 1) {
        if (threadIdx.x < s) red[threadIdx.x] = fmaxf(red[threadIdx.x], red[threadIdx.x + s]);
        __syncthreads();
    }
    m = red[0]; __syncthreads();
    float sum = 0.0f;
    for (int i = threadIdx.x; i < S_RAW; i += 256) {
        float e = expf(src[i] - m);
        attn[(size_t)row * S_RAW + i] = e;
        sum += e;
    }
    red[threadIdx.x] = sum; __syncthreads();
    for (int s = 128; s > 0; s >>= 1) {
        if (threadIdx.x < s) red[threadIdx.x] += red[threadIdx.x + s];
        __syncthreads();
    }
    float inv = 1.0f / red[0];
    for (int i = threadIdx.x; i < S_RAW; i += 256) attn[(size_t)row * S_RAW + i] *= inv;
}

__global__ void rstadd(float* __restrict__ out,
                       const float* __restrict__ attn,
                       const float* __restrict__ vbuf) {
    size_t idx = (size_t)blockIdx.x * blockDim.x + threadIdx.x;
    int col = (int)(idx & 511);
    int h = col >> 7, e = col & 127;
    int s = (int)((idx >> 9) & 1023);
    int b = (int)(idx >> 19);
    out[idx] += attn[((size_t)h * BB + b) * S_RAW + s] * vbuf[(h * BB + b) * DHEAD + e];
}

// ------------------------------------------------------------- launch -----
extern "C" void kernel_launch(void* const* d_in, const int* in_sizes, int n_in,
                              void* d_out, int out_size, void* d_ws, size_t ws_size,
                              hipStream_t stream) {
    (void)in_sizes; (void)n_in; (void)out_size; (void)ws_size;

    const float* in_raw  = (const float*)d_in[0];
    const float* in_sum  = (const float*)d_in[1];
    const int*   len_sum = (const int*)d_in[3];
    const float* Wih[4] = {(const float*)d_in[4],  (const float*)d_in[7],
                           (const float*)d_in[10], (const float*)d_in[13]};
    const float* Whh[4] = {(const float*)d_in[5],  (const float*)d_in[8],
                           (const float*)d_in[11], (const float*)d_in[14]};
    const float* bias[4] = {(const float*)d_in[6],  (const float*)d_in[9],
                            (const float*)d_in[12], (const float*)d_in[15]};
    const float* Wq = (const float*)d_in[16];
    const float* Wk = (const float*)d_in[17];
    const float* Wv = (const float*)d_in[18];

    char* ws = (char*)d_ws;
    size_t off = 0;
    auto alloc = [&](size_t bytes) {
        void* p = ws + off;
        off += (bytes + 255) & ~(size_t)255;
        return p;
    };
    unsigned short* x_raw_bf = (unsigned short*)alloc((size_t)BB * S_RAW * KPAD_IN * 2);
    unsigned short* x_sum_bf = (unsigned short*)alloc((size_t)BB * S_SUM * KPAD_IN * 2);
    unsigned short* wih_bf[4];
    unsigned short* whh_bf[4];
    for (int d = 0; d < 4; ++d) wih_bf[d] = (unsigned short*)alloc((size_t)G4H * KPAD_IN * 2);
    for (int d = 0; d < 4; ++d) whh_bf[d] = (unsigned short*)alloc((size_t)G4H * HH * 2);
    unsigned short* wq_bf = (unsigned short*)alloc((size_t)TWO_H * TWO_H * 2);
    float* xg_rawf = (float*)alloc((size_t)BB * S_RAW * G4H * 4);
    float* xg_rawb = (float*)alloc((size_t)BB * S_RAW * G4H * 4);
    float* xg_sumf = (float*)alloc((size_t)BB * S_SUM * G4H * 4);
    float* xg_sumb = (float*)alloc((size_t)BB * S_SUM * G4H * 4);
    float* out_raw = (float*)alloc((size_t)BB * S_RAW * TWO_H * 4);
    float* out_sum = (float*)alloc((size_t)BB * S_SUM * TWO_H * 4);
    unsigned short* out_raw_bf = (unsigned short*)alloc((size_t)BB * S_RAW * TWO_H * 2);
    float* sum_vec = (float*)alloc((size_t)BB * TWO_H * 4);
    float* kbuf    = (float*)alloc((size_t)NHEAD * BB * DHEAD * 4);
    float* vbuf    = (float*)alloc((size_t)NHEAD * BB * DHEAD * 4);
    float* scores  = (float*)alloc((size_t)NHEAD * BB * S_RAW * 4);
    float* attn    = (float*)alloc((size_t)NHEAD * BB * S_RAW * 4);

    auto cdiv = [](int a, int b) { return (a + b - 1) / b; };

    // 1) pack to bf16
    pack_bf16_pad<<<cdiv(BB * S_RAW * KPAD_IN, 256), 256, 0, stream>>>(in_raw, x_raw_bf, BB * S_RAW, D_IN, KPAD_IN);
    pack_bf16_pad<<<cdiv(BB * S_SUM * KPAD_IN, 256), 256, 0, stream>>>(in_sum, x_sum_bf, BB * S_SUM, D_IN, KPAD_IN);
    for (int d = 0; d < 4; ++d) {
        pack_bf16_pad<<<cdiv(G4H * KPAD_IN, 256), 256, 0, stream>>>(Wih[d], wih_bf[d], G4H, D_IN, KPAD_IN);
        pack_bf16_pad<<<cdiv(G4H * HH, 256), 256, 0, stream>>>(Whh[d], whh_bf[d], G4H, HH, HH);
    }
    pack_wq<<<cdiv(TWO_H * TWO_H, 256), 256, 0, stream>>>(Wq, wq_bf);

    // 2) xg = x @ Wih^T + b  (TDM-fed WMMA GEMMs)  tile: M=128, N=64
    gemm_bf16<<<dim3(BB * S_RAW / 128, G4H / 64), 256, 0, stream>>>(x_raw_bf, wih_bf[0], bias[0], xg_rawf, BB * S_RAW, G4H, KPAD_IN);
    gemm_bf16<<<dim3(BB * S_RAW / 128, G4H / 64), 256, 0, stream>>>(x_raw_bf, wih_bf[1], bias[1], xg_rawb, BB * S_RAW, G4H, KPAD_IN);
    gemm_bf16<<<dim3(BB * S_SUM / 128, G4H / 64), 256, 0, stream>>>(x_sum_bf, wih_bf[2], bias[2], xg_sumf, BB * S_SUM, G4H, KPAD_IN);
    gemm_bf16<<<dim3(BB * S_SUM / 128, G4H / 64), 256, 0, stream>>>(x_sum_bf, wih_bf[3], bias[3], xg_sumb, BB * S_SUM, G4H, KPAD_IN);

    // 3) recurrences: 4 workgroups (raw fwd/bwd, sum fwd/bwd)
    LstmCfg c0{xg_rawf, whh_bf[0], out_raw, S_RAW, 0, 0};
    LstmCfg c1{xg_rawb, whh_bf[1], out_raw, S_RAW, 1, HH};
    LstmCfg c2{xg_sumf, whh_bf[2], out_sum, S_SUM, 0, 0};
    LstmCfg c3{xg_sumb, whh_bf[3], out_sum, S_SUM, 1, HH};
    lstm_scan<<<4, 1024, 176 * 1024, stream>>>(c0, c1, c2, c3);

    // 4) pooling + k/v
    meanpool<<<BB, TWO_H, 0, stream>>>(out_sum, len_sum, sum_vec);
    kvproj<<<cdiv(2 * NHEAD * BB * DHEAD, 256), 256, 0, stream>>>(sum_vec, Wk, Wv, kbuf, vbuf);

    // 5) q projection straight into d_out (concat-head layout)
    pack_bf16_pad<<<cdiv(BB * S_RAW * TWO_H, 256), 256, 0, stream>>>(out_raw, out_raw_bf, BB * S_RAW, TWO_H, TWO_H);
    gemm_bf16<<<dim3(BB * S_RAW / 128, TWO_H / 64), 256, 0, stream>>>(out_raw_bf, wq_bf, (const float*)nullptr, (float*)d_out, BB * S_RAW, TWO_H, TWO_H);

    // 6) attention epilogue
    scoreskern<<<cdiv(NHEAD * BB * S_RAW, 256), 256, 0, stream>>>((const float*)d_out, kbuf, scores);
    softmaxkern<<<NHEAD * BB, 256, 0, stream>>>(scores, attn);
    rstadd<<<cdiv(BB * S_RAW * TWO_H, 256), 256, 0, stream>>>((float*)d_out, attn, vbuf);
}
